// MoELoRADelta_26431228740184
// MI455X (gfx1250) — matrix operand order repaired and checked by reading, not compiled, not run on hardware
//
#include <hip/hip_runtime.h>
#include <hip/hip_bf16.h>
#include <math.h>

typedef __attribute__((ext_vector_type(16))) __bf16 v16bf;
typedef __attribute__((ext_vector_type(8)))  float  v8f;

#define IN_F   4096
#define OUT_F  4096
#define NEXP   4
#define LORA_R 16
#define ER     64          // NEXP * LORA_R
#define NROWS  80          // ER + 16 (Wr rows padded to a full WMMA N-tile)
#define SCALEF 4.0f        // 16 / sqrt(16)
#define CHUNK  128

union ABOp { v16bf v; uint4 q[2]; };

__device__ __forceinline__ unsigned short f2bf(float f) {
  unsigned int u = __float_as_uint(f);
  u += 0x7FFFu + ((u >> 16) & 1u);   // round-to-nearest-even
  return (unsigned short)(u >> 16);
}
// pack two floats into two bf16 halfwords (lo = a, hi = b)
__device__ __forceinline__ unsigned int pack2bf(float a, float b) {
#if __has_builtin(__builtin_amdgcn_cvt_pk_bf16_f32)
  auto p = __builtin_amdgcn_cvt_pk_bf16_f32(a, b);
  unsigned int u;
  __builtin_memcpy(&u, &p, sizeof(u));
  return u;
#else
  return ((unsigned int)f2bf(b) << 16) | (unsigned int)f2bf(a);
#endif
}

// ---------------------------------------------------------------------------
// Kernel 0: AbfX[80][IN_F] bf16 = [ A_all (64 rows) ; Wr (4 rows) ; zeros ]
//           Btbf[o][e*16+r]  bf16  (phase-3 B-operand layout)
// ---------------------------------------------------------------------------
__global__ __launch_bounds__(256)
void moelora_prep_kernel(const float* __restrict__ A, const float* __restrict__ B,
                         const float* __restrict__ Wr,
                         unsigned short* __restrict__ AbfX,
                         unsigned short* __restrict__ Btbf) {
  int g = blockIdx.x * blockDim.x + threadIdx.x;
  if (g < NROWS * IN_F) {
    int row = g >> 12, col = g & (IN_F - 1);
    float v = 0.f;
    if (row < ER)            v = A[g];
    else if (row < ER + NEXP) v = Wr[(size_t)(row - ER) * IN_F + col];
    AbfX[g] = f2bf(v);
  }
  if (g < OUT_F * ER) {
    int o = g >> 6, k = g & 63;
    int e = k >> 4, r = k & 15;
    Btbf[g] = f2bf(B[((size_t)e * OUT_F + o) * LORA_R + r]);
  }
}

// ---------------------------------------------------------------------------
// Kernel 1: block = 16 tokens, 5 waves (160 threads). Wave w computes the
// 16-wide N-tile w of  x_bf16 @ AbfX^T  over the full K:
//   tiles 0-3 -> mid (64 outputs), tile 4 -> router logits (cols 0-3).
// Then top-2 softmax gating and g_mid = gate * SCALE * mid (bf16 out).
// ---------------------------------------------------------------------------
__global__ __launch_bounds__(160)
void moelora_midgate_kernel(const float* __restrict__ x,
                            const unsigned short* __restrict__ AbfX,
                            unsigned short* __restrict__ gmid) {
  __shared__ unsigned int xs[16][CHUNK / 2];   // 4 KB: x chunk as bf16 pairs
  __shared__ float lgt[16][NEXP];              // logits
  __shared__ float gts[16][NEXP];              // gates

  const int tid  = threadIdx.x;
  const int wave = tid >> 5;       // 0..4 == N-tile index
  const int lane = tid & 31;
  const int hi   = lane >> 4;      // lane group (0: lanes 0-15, 1: 16-31)
  const int ln   = lane & 15;
  const int t0   = blockIdx.x * 16;

  // staging assignment (threads 0..127): token sm, 16-wide K slot skl
  const int sm  = tid >> 3;
  const int skl = (tid & 7) * 16;
  const float* xrow = x + (size_t)(t0 + sm) * IN_F + skl;

  v8f acc = {};

  for (int kb = 0; kb < IN_F; kb += CHUNK) {
    if (tid < 128) {
      const float4 v0 = *(const float4*)(xrow + kb);
      const float4 v1 = *(const float4*)(xrow + kb + 4);
      const float4 v2 = *(const float4*)(xrow + kb + 8);
      const float4 v3 = *(const float4*)(xrow + kb + 12);
      if (kb + CHUNK < IN_F) __builtin_prefetch(xrow + kb + CHUNK, 0, 1);
      uint4 p0, p1;
      p0.x = pack2bf(v0.x, v0.y); p0.y = pack2bf(v0.z, v0.w);
      p0.z = pack2bf(v1.x, v1.y); p0.w = pack2bf(v1.z, v1.w);
      p1.x = pack2bf(v2.x, v2.y); p1.y = pack2bf(v2.z, v2.w);
      p1.z = pack2bf(v3.x, v3.y); p1.w = pack2bf(v3.z, v3.w);
      *(uint4*)&xs[sm][(skl >> 1) + 0] = p0;
      *(uint4*)&xs[sm][(skl >> 1) + 4] = p1;
    }
    __syncthreads();

    #pragma unroll
    for (int s = 0; s < 4; s++) {
      const int kk0 = s * 32;
      ABOp a, b;
      // A-operand (16x32 bf16): lane group 0 holds K 0-7 / 16-23,
      // lane group 1 holds K 8-15 / 24-31 — two contiguous 16B LDS loads.
      a.q[0] = *(const uint4*)&xs[ln][(kk0 + 8 * hi) >> 1];
      a.q[1] = *(const uint4*)&xs[ln][(kk0 + 16 + 8 * hi) >> 1];
      // B-operand (32x16 bf16): lanes 0-15 hold K 0-7 / 16-23, lanes 16-31 K 8-15 / 24-31.
      const unsigned short* bp =
          AbfX + (size_t)(wave * 16 + ln) * IN_F + kb + kk0 + hi * 8;
      b.q[0] = *(const uint4*)bp;
      b.q[1] = *(const uint4*)(bp + 16);
      acc = __builtin_amdgcn_wmma_f32_16x16x32_bf16(
          false, a.v, false, b.v, (short)0, acc, false, false);
    }
    __syncthreads();
  }

  // ---- router tile -> logits ----
  if (wave == 4 && ln < NEXP) {
    #pragma unroll
    for (int v = 0; v < 8; v++) lgt[v + 8 * hi][ln] = acc[v];
  }
  __syncthreads();

  // ---- top-2 softmax gates per token ----
  if (tid < 16) {
    float l[NEXP];
    #pragma unroll
    for (int e = 0; e < NEXP; e++) { l[e] = lgt[tid][e]; gts[tid][e] = 0.f; }
    int i1 = 0;
    #pragma unroll
    for (int e = 1; e < NEXP; e++) if (l[e] > l[i1]) i1 = e;
    int i2 = (i1 == 0) ? 1 : 0;
    #pragma unroll
    for (int e = 0; e < NEXP; e++) if (e != i1 && l[e] > l[i2]) i2 = e;
    float g2  = __expf(l[i2] - l[i1]);
    float inv = 1.0f / (1.0f + g2);
    gts[tid][i1] = inv;
    gts[tid][i2] = g2 * inv;
  }
  __syncthreads();

  // ---- apply gate*SCALE, store g_mid as bf16 (N-tile => single expert) ----
  if (wave < 4) {
    #pragma unroll
    for (int v = 0; v < 8; v++) {
      int m   = v + 8 * hi;                 // C/D layout: M = v + 8*(lane/16)
      float g = gts[m][wave] * SCALEF;
      gmid[(size_t)(t0 + m) * ER + wave * 16 + ln] = f2bf(acc[v] * g);
    }
  }
}

// ---------------------------------------------------------------------------
// Kernel 2: delta[t][o] = g_mid[t][:] @ Btbf[o][:]   (K' = 64, 2 WMMA steps)
// Block = 16 tokens x 256 outputs, 8 waves x 2 tiles each.
// ---------------------------------------------------------------------------
__global__ __launch_bounds__(256)
void moelora_delta_kernel(const unsigned short* __restrict__ gmid,
                          const unsigned short* __restrict__ Btbf,
                          float* __restrict__ out) {
  __shared__ unsigned short gs[16][ER];   // 2 KB

  const int tid  = threadIdx.x;
  const int wave = tid >> 5;
  const int lane = tid & 31;
  const int hi   = lane >> 4;
  const int ln   = lane & 15;
  const int t0   = blockIdx.x * 16;
  const int n0   = blockIdx.y * 256;

  // stage g_mid tile (16x64 bf16)
  {
    int i = tid * 4;                 // 1024 halfwords / 256 threads
    int m = i >> 6, k = i & 63;
    *(uint2*)&gs[m][k] = *(const uint2*)&gmid[(size_t)(t0 + m) * ER + k];
  }
  __syncthreads();

  #pragma unroll
  for (int tle = 0; tle < 2; tle++) {
    const int o0 = n0 + (wave * 2 + tle) * 16;
    v8f acc = {};
    #pragma unroll
    for (int s = 0; s < 2; s++) {
      const int kk0 = s * 32;
      ABOp a, b;
      a.q[0] = *(const uint4*)&gs[ln][kk0 + 8 * hi];
      a.q[1] = *(const uint4*)&gs[ln][kk0 + 16 + 8 * hi];
      const unsigned short* bp = Btbf + (size_t)(o0 + ln) * ER + kk0 + hi * 8;
      b.q[0] = *(const uint4*)bp;
      b.q[1] = *(const uint4*)(bp + 16);
      acc = __builtin_amdgcn_wmma_f32_16x16x32_bf16(
          false, a.v, false, b.v, (short)0, acc, false, false);
    }
    #pragma unroll
    for (int v = 0; v < 8; v++) {
      out[(size_t)(t0 + v + 8 * hi) * OUT_F + o0 + ln] = acc[v];
    }
  }
}

// ---------------------------------------------------------------------------
extern "C" void kernel_launch(void* const* d_in, const int* in_sizes, int n_in,
                              void* d_out, int out_size, void* d_ws, size_t ws_size,
                              hipStream_t stream) {
  const float* x  = (const float*)d_in[0];
  const float* A  = (const float*)d_in[1];
  const float* B  = (const float*)d_in[2];
  const float* Wr = (const float*)d_in[3];
  float* out = (float*)d_out;

  const int T = in_sizes[0] / IN_F;   // 4096 tokens

  unsigned short* AbfX = (unsigned short*)d_ws;                 // 640 KB
  unsigned short* Btbf = AbfX + (size_t)NROWS * IN_F;           // 512 KB
  unsigned short* gmid = Btbf + (size_t)OUT_F * ER;             // T*64*2 B

  moelora_prep_kernel<<<(NROWS * IN_F + 255) / 256, 256, 0, stream>>>(
      A, B, Wr, AbfX, Btbf);
  moelora_midgate_kernel<<<T / 16, 160, 0, stream>>>(x, AbfX, gmid);
  moelora_delta_kernel<<<dim3(T / 16, OUT_F / 256), 256, 0, stream>>>(
      gmid, Btbf, out);
}
